// Loss_SSIM_45578192945789
// MI455X (gfx1250) — matrix operand order repaired and checked by reading, not compiled, not run on hardware
//
#include <hip/hip_runtime.h>

// CDNA5 / gfx1250 SSIM via WMMA.
// Separable 11x11 Gaussian conv expressed as two banded 16x16x32 f16 matmuls
// per 16x16 tile per stream; 5 streams (x1, x2, x1^2, x2^2, x1*x2);
// elementwise SSIM combine on the f32 accumulators; deterministic two-stage
// mean reduction. Round 2: wave-uniform interior fast path (no per-element
// predication; loads fold to base+imm-offset) and compile-time-packed f16
// weight fragments in __constant__ memory.

typedef __attribute__((ext_vector_type(16))) _Float16 v16h;
typedef __attribute__((ext_vector_type(8)))  float    v8f;

#define IMG_H 512
#define IMG_W 512
#define NPLANES 48                 // 16 batch * 3 channels
#define TILES_PER_PLANE 1024       // 32 x 32 tiles of 16x16
#define NTILES (NPLANES * TILES_PER_PLANE)   // 49152
#define WAVES_PER_BLOCK 4
#define NBLOCKS (NTILES / WAVES_PER_BLOCK)   // 12288
#define TSTRIDE 36                 // LDS row stride (floats): 16B aligned, conflict-free

// ---------------- compile-time weight fragment tables ----------------------
// Normalized 11-tap Gaussian, sigma = 1.5 (matches reference).
constexpr double G_TAP[11] = {
    0.00102838, 0.00759876, 0.03600077, 0.10936070, 0.21300554,
    0.26601173,
    0.21300554, 0.10936070, 0.03600077, 0.00759876, 0.00102838
};

// constexpr float -> IEEE f16 bits, round-to-nearest-even (inputs are zero or
// normal-range positives, so no denorm/inf handling needed).
constexpr unsigned short f2h(double f) {
    if (f == 0.0) return 0;
    int e = 0;
    double m = f;
    while (m < 1.0)  { m *= 2.0; --e; }
    while (m >= 2.0) { m /= 2.0; ++e; }
    double frac = (m - 1.0) * 1024.0;
    int fi = (int)frac;
    double rem = frac - (double)fi;
    if (rem > 0.5 || (rem == 0.5 && (fi & 1))) ++fi;
    if (fi == 1024) { fi = 0; ++e; }
    return (unsigned short)(((e + 15) << 10) | fi);
}

struct WTab {
    alignas(32) unsigned int wv[32][8];   // A-fragment (Wv) per lane, packed f16x2
    alignas(32) unsigned int wh[32][8];   // B-fragment (Wh) per lane, packed f16x2
};

constexpr WTab make_wtab() {
    WTab t{};
    for (int lane = 0; lane < 32; ++lane) {
        const int half = lane >> 4;
        const int M    = lane & 15;      // A row / B col for this lane
        unsigned short hv[16] = {};
        unsigned short hh[16] = {};
        for (int e = 0; e < 16; ++e) {
            // A operand, contiguous k order chosen here: row M=lane&15, K = e + 16*half (stage-1 B matches)
            const int K = e + 16 * half;
            const int d = K - M;
            hv[e] = f2h((d >= 0 && d <= 10) ? G_TAP[d] : 0.0);
            // B layout (32x16 f16): lane holds col N=lane&15; lanes<16: K=e (e<8) / e+8 ; lanes>=16: K=e+8 / e+16
            const int Kh = (half == 0) ? ((e < 8) ? e : e + 8)
                                       : ((e < 8) ? e + 8 : e + 16);
            const int dh = Kh - M;
            hh[e] = f2h((dh >= 0 && dh <= 10) ? G_TAP[dh] : 0.0);
        }
        for (int e = 0; e < 8; ++e) {
            t.wv[lane][e] = (unsigned)hv[2*e] | ((unsigned)hv[2*e + 1] << 16);
            t.wh[lane][e] = (unsigned)hh[2*e] | ((unsigned)hh[2*e + 1] << 16);
        }
    }
    return t;
}

__device__ __constant__ const WTab c_wtab = make_wtab();

// ---------------------------------------------------------------------------
__global__ __launch_bounds__(128) void ssim_tiles_kernel(
    const float* __restrict__ img1, const float* __restrict__ img2,
    float* __restrict__ partials)
{
    __shared__ float Tlds[WAVES_PER_BLOCK][5][16][TSTRIDE];  // 46080 B
    __shared__ float wsum[WAVES_PER_BLOCK];

    const int tid  = threadIdx.x;
    const int lane = tid & 31;
    const int wave = tid >> 5;
    const int tile = blockIdx.x * WAVES_PER_BLOCK + wave;

    const int plane = tile >> 10;
    const int twin  = tile & 1023;
    const int R0 = (twin >> 5) << 4;      // tile row origin
    const int C0 = (twin & 31) << 4;      // tile col origin
    const size_t pbase = (size_t)plane * (IMG_H * IMG_W);

    const int n    = lane & 15;
    const int half = lane >> 4;
    const int M    = lane & 15;

    // Weight fragments: 2x global_load_b128 each from __constant__.
    const v16h Wv = *reinterpret_cast<const v16h*>(c_wtab.wv[lane]);
    const v16h Wh = *reinterpret_cast<const v16h*>(c_wtab.wh[lane]);

    // Wave-uniform interior test: rows/cols touched are [R0-5, R0+26] x [C0-5, C0+26].
    const bool interior = (R0 >= 16) && (R0 <= IMG_H - 32) &&
                          (C0 >= 16) && (C0 <= IMG_W - 32);

    // ---- Stage 1: T[i][u] = sum_k g[k-i] * X[R0-5+k][C0-5+u], two 16-col tiles ----
    #pragma unroll
    for (int t = 0; t < 2; ++t) {
        float x1v[16], x2v[16];
        const int col = C0 - 5 + 16 * t + n;
        const int row0 = R0 - 5 + 16 * half;          // k = 16*half + e (B layout)
        if (interior) {
            // Fast path: one address, 16 loads with immediate offsets, no masks.
            const float* p1 = img1 + (pbase + (size_t)row0 * IMG_W + col);
            const float* p2 = img2 + (pbase + (size_t)row0 * IMG_W + col);
            #pragma unroll
            for (int e = 0; e < 16; ++e) {
                x1v[e] = p1[(size_t)e * IMG_W];
                x2v[e] = p2[(size_t)e * IMG_W];
            }
        } else {
            const bool colOK = (col >= 0) && (col < IMG_W);
            #pragma unroll
            for (int e = 0; e < 16; ++e) {
                const int row = row0 + e;
                const bool ok = colOK && (row >= 0) && (row < IMG_H);
                const size_t idx = ok ? (pbase + (size_t)row * IMG_W + col) : pbase;
                const float a = img1[idx];
                const float b = img2[idx];
                x1v[e] = ok ? a : 0.0f;
                x2v[e] = ok ? b : 0.0f;
            }
        }
        #pragma unroll
        for (int s = 0; s < 5; ++s) {
            v16h B;
            #pragma unroll
            for (int e = 0; e < 16; ++e) {
                float v;
                if      (s == 0) v = x1v[e];
                else if (s == 1) v = x2v[e];
                else if (s == 2) v = x1v[e] * x1v[e];
                else if (s == 3) v = x2v[e] * x2v[e];
                else             v = x1v[e] * x2v[e];
                B[e] = (_Float16)v;
            }
            v8f D = {};
            D = __builtin_amdgcn_wmma_f32_16x16x32_f16(
                    false, Wv, false, B, (short)0, D, false, false);
            // D layout -> T rows: lane holds (M = v + 8*half, u = n + 16*t)
            #pragma unroll
            for (int v = 0; v < 8; ++v)
                Tlds[wave][s][v + 8 * half][n + 16 * t] = D[v];
        }
    }

    __syncthreads();

    // ---- Stage 2: Out[i][j] = sum_u T[i][u] * g[u-j]  (A = T, B = Wh) ----
    v8f acc[5];
    {
        const int u0 = half ? 8 : 0;
        const int u1 = half ? 24 : 16;
        #pragma unroll
        for (int s = 0; s < 5; ++s) {
            v16h A;
            #pragma unroll
            for (int e = 0; e < 8; ++e) {
                A[e]     = (_Float16)Tlds[wave][s][M][u0 + e];
                A[e + 8] = (_Float16)Tlds[wave][s][M][u1 + e];
            }
            v8f D = {};
            D = __builtin_amdgcn_wmma_f32_16x16x32_f16(
                    false, A, false, Wh, (short)0, D, false, false);
            acc[s] = D;
        }
    }

    // ---- Elementwise SSIM (all 5 accumulators share the C/D lane layout) ----
    const float C1 = 0.0001f;   // 0.01^2
    const float C2 = 0.0009f;   // 0.03^2
    float lsum = 0.0f;
    #pragma unroll
    for (int v = 0; v < 8; ++v) {
        const float mu1 = acc[0][v], mu2 = acc[1][v];
        const float s11 = acc[2][v], s22 = acc[3][v], s12 = acc[4][v];
        const float mu1s = mu1 * mu1, mu2s = mu2 * mu2, mu12 = mu1 * mu2;
        const float sig1 = s11 - mu1s;
        const float sig2 = s22 - mu2s;
        const float cs12 = s12 - mu12;
        const float num = (2.0f * mu12 + C1) * (2.0f * cs12 + C2);
        const float den = (mu1s + mu2s + C1) * (sig1 + sig2 + C2);
        lsum += num / den;
    }

    // ---- wave32 reduction, then per-block partial (deterministic) ----
    #pragma unroll
    for (int off = 16; off > 0; off >>= 1)
        lsum += __shfl_xor(lsum, off, 32);
    if (lane == 0) wsum[wave] = lsum;
    __syncthreads();
    if (tid == 0)
        partials[blockIdx.x] = wsum[0] + wsum[1] + wsum[2] + wsum[3];
}

__global__ __launch_bounds__(256) void ssim_reduce_kernel(
    const float* __restrict__ partials, int n, float* __restrict__ out)
{
    __shared__ float sdata[256];
    float s = 0.0f;
    for (int i = threadIdx.x; i < n; i += 256)   // fixed order -> deterministic
        s += partials[i];
    sdata[threadIdx.x] = s;
    __syncthreads();
    #pragma unroll
    for (int off = 128; off > 0; off >>= 1) {
        if ((int)threadIdx.x < off) sdata[threadIdx.x] += sdata[threadIdx.x + off];
        __syncthreads();
    }
    if (threadIdx.x == 0)
        out[0] = sdata[0] * (1.0f / (float)(NPLANES * IMG_H * IMG_W));
}

extern "C" void kernel_launch(void* const* d_in, const int* in_sizes, int n_in,
                              void* d_out, int out_size, void* d_ws, size_t ws_size,
                              hipStream_t stream)
{
    const float* img1 = (const float*)d_in[0];
    const float* img2 = (const float*)d_in[1];
    float* out      = (float*)d_out;
    float* partials = (float*)d_ws;   // NBLOCKS floats = 48 KB

    ssim_tiles_kernel<<<dim3(NBLOCKS), dim3(128), 0, stream>>>(img1, img2, partials);
    ssim_reduce_kernel<<<dim3(1), dim3(256), 0, stream>>>(partials, NBLOCKS, out);
}
